// PointNetSAModuleMSG_71691594104933
// MI455X (gfx1250) — compile-verified, matrix operand order checked by reading, not run on hardware
//
#include <hip/hip_runtime.h>
#include <hip/hip_bf16.h>

typedef __attribute__((ext_vector_type(16))) _Float16 v16h;
typedef __attribute__((ext_vector_type(8)))  _Float16 v8h;
typedef __attribute__((ext_vector_type(8)))  float    v8f;

#define NPTS   16384
#define NBATCH 8
#define MCENT  1024
#define FEATCH 320

// ---------------------------------------------------------------- utilities
__global__ void zero_kernel(float* __restrict__ p, int n) {
  for (int i = blockIdx.x * blockDim.x + threadIdx.x; i < n; i += gridDim.x * blockDim.x)
    p[i] = 0.f;
}

// ---------------------------------------------------------------- FPS
__global__ __launch_bounds__(1024) void fps_kernel(const float* __restrict__ xyz,
                                                   float* __restrict__ newxyz) {
  const int b   = blockIdx.x;
  const int tid = threadIdx.x;
  const float* X = xyz + (size_t)b * NPTS * 3;

  float px[16], py[16], pz[16], md[16];
  const float x0 = X[0], y0 = X[1], z0 = X[2];
#pragma unroll
  for (int j = 0; j < 16; ++j) {
    const int p = tid + 1024 * j;
    const float x = X[p * 3 + 0], y = X[p * 3 + 1], z = X[p * 3 + 2];
    px[j] = x; py[j] = y; pz[j] = z;
    const float dx = x - x0, dy = y - y0, dz = z - z0;
    md[j] = dx * dx + dy * dy + dz * dz;
  }

  __shared__ float s_val[32];
  __shared__ int   s_idx[32];
  __shared__ float s_c[3];

  if (tid == 0) {
    float* o = newxyz + (size_t)b * MCENT * 3;
    o[0] = x0; o[1] = y0; o[2] = z0;
  }

  const int lane = tid & 31, wid = tid >> 5;
  for (int it = 1; it < MCENT; ++it) {
    float bv = -1.f; int bi = 0;
#pragma unroll
    for (int j = 0; j < 16; ++j)
      if (md[j] > bv) { bv = md[j]; bi = tid + 1024 * j; }
#pragma unroll
    for (int off = 16; off > 0; off >>= 1) {
      const float ov = __shfl_xor(bv, off, 32);
      const int   oi = __shfl_xor(bi, off, 32);
      if (ov > bv || (ov == bv && oi < bi)) { bv = ov; bi = oi; }
    }
    if (lane == 0) { s_val[wid] = bv; s_idx[wid] = bi; }
    __syncthreads();
    if (tid == 0) {
      float best = s_val[0]; int besti = s_idx[0];
      for (int w2 = 1; w2 < 32; ++w2)
        if (s_val[w2] > best || (s_val[w2] == best && s_idx[w2] < besti)) {
          best = s_val[w2]; besti = s_idx[w2];
        }
      const float cx = X[besti * 3], cy = X[besti * 3 + 1], cz = X[besti * 3 + 2];
      s_c[0] = cx; s_c[1] = cy; s_c[2] = cz;
      float* o = newxyz + ((size_t)b * MCENT + it) * 3;
      o[0] = cx; o[1] = cy; o[2] = cz;
    }
    __syncthreads();
    const float cx = s_c[0], cy = s_c[1], cz = s_c[2];
#pragma unroll
    for (int j = 0; j < 16; ++j) {
      const float dx = px[j] - cx, dy = py[j] - cy, dz = pz[j] - cz;
      md[j] = fminf(md[j], dx * dx + dy * dy + dz * dz);
    }
  }
}

// ---------------------------------------------------------------- ball query
template <int NS>
__global__ __launch_bounds__(256) void ballq_kernel(const float* __restrict__ xyz,
                                                    const float* __restrict__ newxyz,
                                                    int* __restrict__ idx, float r2) {
  const int gw   = (blockIdx.x * blockDim.x + threadIdx.x) >> 5;
  const int lane = threadIdx.x & 31;
  if (gw >= NBATCH * MCENT) return;
  const int b = gw / MCENT;
  const float* X = xyz + (size_t)b * NPTS * 3;
  const float* c = newxyz + (size_t)gw * 3;
  const float cx = c[0], cy = c[1], cz = c[2];
  int* out = idx + (size_t)gw * NS;

  int cnt = 0, first = -1;
  for (int base = 0; base < NPTS && cnt < NS; base += 32) {
    const int p = base + lane;
    const float dx = X[p * 3] - cx, dy = X[p * 3 + 1] - cy, dz = X[p * 3 + 2] - cz;
    const bool pred = (dx * dx + dy * dy + dz * dz) < r2;
    const unsigned mask = (unsigned)__ballot(pred);
    if (first < 0 && mask) first = base + __builtin_ctz(mask);
    if (pred) {
      const int rank = cnt + __popc(mask & ((1u << lane) - 1u));
      if (rank < NS) out[rank] = p;
    }
    cnt += __popc(mask);
  }
  if (cnt > NS) cnt = NS;
  const int f = (first < 0) ? 0 : first;
  for (int r = cnt + lane; r < NS; r += 32) out[r] = f;
}

// ---------------------------------------------------------------- weight packing
// Pack W (K x COUT, row-major f32) into per-lane A-fragment order so each
// WMMA A fragment is one aligned 32B LDS read.
template <int K, int COUT>
__device__ inline void pack_w(const float* __restrict__ W, _Float16* __restrict__ dst,
                              int kreal, int tid) {
  for (int i = tid; i < K * COUT; i += 128) {
    const int p  = i & 31;
    const int t  = i >> 5;
    const int ch = t % COUT;
    const int kb = (t / COUT) * 32;
    const int h  = p >> 4, q = p & 15;
    const int relk = (q < 8) ? (8 * h + q) : (16 + 8 * h + (q - 8));
    const int k = kb + relk;
    dst[i] = (k < kreal) ? (_Float16)W[(size_t)k * COUT + ch] : (_Float16)0.f;
  }
}

// ---------------------------------------------------------------- WMMA tile
// D = A x B : A = packed weights (16 channels x K), B = activations
// (K x 16 points) read straight out of the row-major x-buffer.
template <int K, int COUT, int LD>
__device__ inline v8f wmma_tileT(const _Float16* __restrict__ wp,
                                 const _Float16* __restrict__ x, int ct, int lane) {
  const int half = (lane >> 4) & 1;
  const int pc   = lane & 15;
  v8f acc = {0.f, 0.f, 0.f, 0.f, 0.f, 0.f, 0.f, 0.f};
#pragma unroll
  for (int kb = 0; kb < K; kb += 32) {
    const v16h a = *(const v16h*)(wp + (((kb >> 5) * COUT + ct * 16 + pc) << 5) +
                                  half * 16);
    const v8h blo = *(const v8h*)(x + pc * LD + kb + 8 * half);
    const v8h bhi = *(const v8h*)(x + pc * LD + kb + 16 + 8 * half);
    const v16h b = __builtin_shufflevector(blo, bhi, 0, 1, 2, 3, 4, 5, 6, 7, 8, 9,
                                           10, 11, 12, 13, 14, 15);
    acc = __builtin_amdgcn_wmma_f32_16x16x32_f16(false, a, false, b, (short)0, acc,
                                                 false, false);
  }
  return acc;
}

// ---------------------------------------------------------------- fused MLP pass
// PASS=1..3: recompute layers 1..PASS per 16-point tile in LDS; accumulate
//            channel sum/sumsq of layer-PASS pre-BN output (for BN stats).
// PASS=4:    full chain + BN+ReLU + atomicMax neighborhood pooling to output.
// BN+bias are folded into 2 per-channel constants staged in LDS once per block.
template <int NS, int C1, int C2, int C3, int PASS, int COFF>
__global__ __launch_bounds__(128) void mlp_pass_kernel(
    const float* __restrict__ xyz, const float* __restrict__ points,
    const float* __restrict__ newxyz, const int* __restrict__ idx,
    const float* __restrict__ W1, const float* __restrict__ B1,
    const float* __restrict__ W2, const float* __restrict__ B2,
    const float* __restrict__ W3, const float* __restrict__ B3,
    const float* __restrict__ bn1, const float* __restrict__ bn2,
    const float* __restrict__ bn3, float* __restrict__ sums,
    float* __restrict__ outFeat) {
  constexpr int KM1    = (C1 > 32) ? C1 : 32;
  constexpr int KMAX   = (C2 > KM1) ? C2 : KM1;
  constexpr int NTILES = NBATCH * MCENT * NS / 16;
  constexpr int CP     = (PASS == 1) ? C1 : ((PASS == 2) ? C2 : C3);

  __shared__ __align__(32) _Float16 sW1[32 * C1];
  __shared__ __align__(32) _Float16 sW2[(PASS >= 2) ? C1 * C2 : 32];
  __shared__ __align__(32) _Float16 sW3[(PASS >= 3) ? C2 * C3 : 32];
  __shared__ __align__(32) _Float16 sX0[16 * KMAX];
  __shared__ __align__(32) _Float16 sX1[16 * KMAX];
  // Folded per-channel constants: [C] = scale, [C..2C) = scale*bias + shift.
  __shared__ __align__(32) float sK1[(PASS >= 2) ? 2 * C1 : 8];
  __shared__ __align__(32) float sK2[(PASS >= 3) ? 2 * C2 : 8];
  __shared__ __align__(32) float sK3[(PASS == 4) ? 2 * C3 : 8];
  __shared__ __align__(32) float sBias[(PASS <= 3) ? CP : 8];

  const int tid = threadIdx.x;
  const int wave = tid >> 5, lane = tid & 31;
  const int half = (lane >> 4) & 1, pc = lane & 15;

  pack_w<32, C1>(W1, sW1, 9, tid);
  if constexpr (PASS >= 2) pack_w<C1, C2>(W2, sW2, C1, tid);
  if constexpr (PASS >= 3) pack_w<C2, C3>(W3, sW3, C2, tid);
  if constexpr (PASS >= 2)
    for (int i = tid; i < C1; i += 128) {
      const float sc = bn1[i];
      sK1[i] = sc; sK1[C1 + i] = sc * B1[i] + bn1[C1 + i];
    }
  if constexpr (PASS >= 3)
    for (int i = tid; i < C2; i += 128) {
      const float sc = bn2[i];
      sK2[i] = sc; sK2[C2 + i] = sc * B2[i] + bn2[C2 + i];
    }
  if constexpr (PASS == 4)
    for (int i = tid; i < C3; i += 128) {
      const float sc = bn3[i];
      sK3[i] = sc; sK3[C3 + i] = sc * B3[i] + bn3[C3 + i];
    }
  if constexpr (PASS <= 3) {
    const float* Bp = (PASS == 1) ? B1 : ((PASS == 2) ? B2 : B3);
    for (int i = tid; i < CP; i += 128) sBias[i] = Bp[i];
  }
  __syncthreads();

  float ps[2][8], psq[2][8];
#pragma unroll
  for (int t = 0; t < 2; ++t)
#pragma unroll
    for (int r = 0; r < 8; ++r) { ps[t][r] = 0.f; psq[t][r] = 0.f; }

  for (int tile = blockIdx.x; tile < NTILES; tile += gridDim.x) {
    // Gather 16 grouped points: [points(6) | rel_xyz(3) | zero-pad to 32].
    if (tid < 16) {
      const int g  = tile * 16 + tid;
      const int bm = g / NS, s = g - bm * NS;
      const int bb = bm / MCENT;
      const int p  = idx[(size_t)bm * NS + s];
      const float* pf  = points + ((size_t)bb * NPTS + p) * 6;
      const float* pxp = xyz + ((size_t)bb * NPTS + p) * 3;
      const float* cc  = newxyz + (size_t)bm * 3;
      v8h r0, r1;
#pragma unroll
      for (int c = 0; c < 6; ++c) r0[c] = (_Float16)pf[c];
      r0[6] = (_Float16)(pxp[0] - cc[0]);
      r0[7] = (_Float16)(pxp[1] - cc[1]);
      r1[0] = (_Float16)(pxp[2] - cc[2]);
#pragma unroll
      for (int c = 1; c < 8; ++c) r1[c] = (_Float16)0.f;
      const v8h rz = {(_Float16)0.f, (_Float16)0.f, (_Float16)0.f, (_Float16)0.f,
                      (_Float16)0.f, (_Float16)0.f, (_Float16)0.f, (_Float16)0.f};
      _Float16* xr = sX0 + tid * KMAX;
      *(v8h*)(xr + 0)  = r0;
      *(v8h*)(xr + 8)  = r1;
      *(v8h*)(xr + 16) = rz;
      *(v8h*)(xr + 24) = rz;
    }
    __syncthreads();

    // ---- layer 1 ----
#pragma unroll
    for (int ct = wave; ct < C1 / 16; ct += 4) {
      v8f y = wmma_tileT<32, C1, KMAX>(sW1, sX0, ct, lane);
      const int chb = ct * 16 + 8 * half;
      const int ti  = (ct - wave) >> 2;
      if constexpr (PASS == 1) {
#pragma unroll
        for (int r = 0; r < 8; ++r) {
          const float v = y[r] + sBias[chb + r];
          ps[ti][r] += v; psq[ti][r] += v * v;
        }
      } else {
        v8h o;
#pragma unroll
        for (int r = 0; r < 8; ++r) {
          const int ch = chb + r;
          const float v = sK1[ch] * y[r] + sK1[C1 + ch];
          o[r] = (_Float16)fmaxf(v, 0.f);
        }
        *(v8h*)(sX1 + pc * KMAX + chb) = o;
      }
    }
    __syncthreads();

    // ---- layer 2 ----
    if constexpr (PASS >= 2) {
#pragma unroll
      for (int ct = wave; ct < C2 / 16; ct += 4) {
        v8f y = wmma_tileT<C1, C2, KMAX>(sW2, sX1, ct, lane);
        const int chb = ct * 16 + 8 * half;
        const int ti  = (ct - wave) >> 2;
        if constexpr (PASS == 2) {
#pragma unroll
          for (int r = 0; r < 8; ++r) {
            const float v = y[r] + sBias[chb + r];
            ps[ti][r] += v; psq[ti][r] += v * v;
          }
        } else {
          v8h o;
#pragma unroll
          for (int r = 0; r < 8; ++r) {
            const int ch = chb + r;
            const float v = sK2[ch] * y[r] + sK2[C2 + ch];
            o[r] = (_Float16)fmaxf(v, 0.f);
          }
          *(v8h*)(sX0 + pc * KMAX + chb) = o;
        }
      }
      __syncthreads();
    }

    // ---- layer 3 ----
    if constexpr (PASS >= 3) {
#pragma unroll
      for (int ct = wave; ct < C3 / 16; ct += 4) {
        v8f y = wmma_tileT<C2, C3, KMAX>(sW3, sX0, ct, lane);
        const int chb = ct * 16 + 8 * half;
        const int ti  = (ct - wave) >> 2;
        if constexpr (PASS == 3) {
#pragma unroll
          for (int r = 0; r < 8; ++r) {
            const float v = y[r] + sBias[chb + r];
            ps[ti][r] += v; psq[ti][r] += v * v;
          }
        } else {
          const int g  = tile * 16 + pc;
          const int bm = g / NS;
          float* orow = outFeat + (size_t)bm * FEATCH + COFF + chb;
#pragma unroll
          for (int r = 0; r < 8; ++r) {
            const int ch = chb + r;
            float v = sK3[ch] * y[r] + sK3[C3 + ch];
            v = fmaxf(v, 0.f);
            atomicMax((int*)(orow + r), __float_as_int(v));
          }
        }
      }
      __syncthreads();
    }
  }

  // Flush stats: butterfly-reduce the 16 point-lanes of each half first so
  // only lane pc==0 touches the global accumulators.
  if constexpr (PASS <= 3) {
#pragma unroll
    for (int t = 0; t < 2; ++t)
#pragma unroll
      for (int r = 0; r < 8; ++r)
#pragma unroll
        for (int off = 1; off < 16; off <<= 1) {
          ps[t][r]  += __shfl_xor(ps[t][r], off, 32);
          psq[t][r] += __shfl_xor(psq[t][r], off, 32);
        }
    if (pc == 0) {
#pragma unroll
      for (int ct = wave; ct < CP / 16; ct += 4) {
        const int ti  = (ct - wave) >> 2;
        const int chb = ct * 16 + 8 * half;
#pragma unroll
        for (int r = 0; r < 8; ++r) {
          atomicAdd(sums + chb + r, ps[ti][r]);
          atomicAdd(sums + CP + chb + r, psq[ti][r]);
        }
      }
    }
  }
}

// ---------------------------------------------------------------- BN fold
__global__ void bn_stats_kernel(const float* __restrict__ sums,
                                const float* __restrict__ gamma,
                                const float* __restrict__ beta,
                                float* __restrict__ bn, int C, float invN) {
  const int c = threadIdx.x;
  if (c >= C) return;
  const float mean = sums[c] * invN;
  const float var  = sums[C + c] * invN - mean * mean;
  const float sc   = gamma[c] * rsqrtf(var + 1e-5f);
  bn[c]     = sc;
  bn[C + c] = beta[c] - mean * sc;
}

// ---------------------------------------------------------------- per-scale driver
template <int NS, int C1, int C2, int C3, int COFF>
static void run_scale(const float* xyz, const float* pts, const float* nx,
                      const int* idx, const float* const* prm, float* sums,
                      float* bn, float* outFeat, hipStream_t st) {
  const float *W1 = prm[0], *B1 = prm[1], *G1 = prm[2], *Be1 = prm[3];
  const float *W2 = prm[4], *B2 = prm[5], *G2 = prm[6], *Be2 = prm[7];
  const float *W3 = prm[8], *B3 = prm[9], *G3 = prm[10], *Be3 = prm[11];
  const float invN = 1.f / (float)((size_t)NBATCH * MCENT * NS);
  float *s1 = sums, *s2 = sums + 256, *s3 = sums + 512;
  float *b1 = bn, *b2 = bn + 256, *b3 = bn + 512;
  dim3 grid(512), blk(128);

  mlp_pass_kernel<NS, C1, C2, C3, 1, COFF><<<grid, blk, 0, st>>>(
      xyz, pts, nx, idx, W1, B1, W2, B2, W3, B3, b1, b2, b3, s1, outFeat);
  bn_stats_kernel<<<1, 128, 0, st>>>(s1, G1, Be1, b1, C1, invN);
  mlp_pass_kernel<NS, C1, C2, C3, 2, COFF><<<grid, blk, 0, st>>>(
      xyz, pts, nx, idx, W1, B1, W2, B2, W3, B3, b1, b2, b3, s2, outFeat);
  bn_stats_kernel<<<1, 128, 0, st>>>(s2, G2, Be2, b2, C2, invN);
  mlp_pass_kernel<NS, C1, C2, C3, 3, COFF><<<grid, blk, 0, st>>>(
      xyz, pts, nx, idx, W1, B1, W2, B2, W3, B3, b1, b2, b3, s3, outFeat);
  bn_stats_kernel<<<1, 128, 0, st>>>(s3, G3, Be3, b3, C3, invN);
  mlp_pass_kernel<NS, C1, C2, C3, 4, COFF><<<grid, blk, 0, st>>>(
      xyz, pts, nx, idx, W1, B1, W2, B2, W3, B3, b1, b2, b3, s3, outFeat);
}

// ---------------------------------------------------------------- entry point
extern "C" void kernel_launch(void* const* d_in, const int* in_sizes, int n_in,
                              void* d_out, int out_size, void* d_ws, size_t ws_size,
                              hipStream_t stream) {
  const float* xyz = (const float*)d_in[0];
  const float* pts = (const float*)d_in[1];
  const float* prm0[12]; const float* prm1[12]; const float* prm2[12];
  for (int i = 0; i < 12; ++i) {
    prm0[i] = (const float*)d_in[2 + i];
    prm1[i] = (const float*)d_in[14 + i];
    prm2[i] = (const float*)d_in[26 + i];
  }

  float* newxyz  = (float*)d_out;                       // [8,1024,3]
  float* outFeat = newxyz + (size_t)NBATCH * MCENT * 3; // [8,1024,320]

  char* wp = (char*)d_ws;
  int* idx0 = (int*)wp; wp += (size_t)NBATCH * MCENT * 16 * sizeof(int);
  int* idx1 = (int*)wp; wp += (size_t)NBATCH * MCENT * 32 * sizeof(int);
  int* idx2 = (int*)wp; wp += (size_t)NBATCH * MCENT * 128 * sizeof(int);
  float* sumsAll = (float*)wp; wp += 9 * 256 * sizeof(float);
  float* bnAll   = (float*)wp; wp += 9 * 256 * sizeof(float);

  zero_kernel<<<512, 256, 0, stream>>>(outFeat, NBATCH * MCENT * FEATCH);
  zero_kernel<<<4, 256, 0, stream>>>(sumsAll, 9 * 256);

  fps_kernel<<<NBATCH, 1024, 0, stream>>>(xyz, newxyz);

  ballq_kernel<16><<<NBATCH * MCENT / 8, 256, 0, stream>>>(xyz, newxyz, idx0, 0.01f);
  ballq_kernel<32><<<NBATCH * MCENT / 8, 256, 0, stream>>>(xyz, newxyz, idx1, 0.04f);
  ballq_kernel<128><<<NBATCH * MCENT / 8, 256, 0, stream>>>(xyz, newxyz, idx2, 0.16f);

  run_scale<16, 32, 32, 64, 0>(xyz, pts, newxyz, idx0, prm0, sumsAll + 0 * 768,
                               bnAll + 0 * 768, outFeat, stream);
  run_scale<32, 64, 64, 128, 64>(xyz, pts, newxyz, idx1, prm1, sumsAll + 1 * 768,
                                 bnAll + 1 * 768, outFeat, stream);
  run_scale<128, 64, 96, 128, 192>(xyz, pts, newxyz, idx2, prm2, sumsAll + 2 * 768,
                                   bnAll + 2 * 768, outFeat, stream);
}
